// BiMambaBlock_4793183502528
// MI455X (gfx1250) — matrix operand order repaired and checked by reading, not compiled, not run on hardware
//
#include <hip/hip_runtime.h>
#include <cstdint>
#include <cstddef>

#define DEV __device__ __forceinline__

typedef __attribute__((ext_vector_type(16))) __bf16 v16bf;
typedef __attribute__((ext_vector_type(8)))  float  v8f;

namespace cfg {
constexpr int D  = 1024;
constexpr int E  = 2048;
constexpr int NS = 64;          // state dim N
constexpr int R  = 64;          // dt rank
constexpr int KT = 4;           // conv taps
constexpr int B  = 2;
constexpr int L  = 2048;
constexpr int M  = B * L;       // 4096 rows
constexpr int E2 = 2 * E;       // 4096
constexpr int XW = R + 2 * NS;  // 192
constexpr float EPS = 1e-5f;
}

struct __align__(16) U4 { unsigned x, y, z, w; };

DEV __bf16 to_bf16(float f) {
  unsigned u = __builtin_bit_cast(unsigned, f);
  u += 0x7fffu + ((u >> 16) & 1u);               // round-to-nearest-even
  unsigned short s = (unsigned short)(u >> 16);
  return __builtin_bit_cast(__bf16, s);
}

DEV float sigmoidf_(float x) { return 1.f / (1.f + __expf(-x)); }

// ---------------- WMMA fragment loaders (bf16 NT GEMM) ----------------
// A fragment: 16(M) x 32(K). Lane l<16 -> row l, K {0..7,16..23};
// lane l>=16 -> row l-16, K {8..15,24..31}.  (ISA 7.12.2, 16-bit A 16x32)
DEV v16bf load_a(const __bf16* A, int lda, int lane) {
  const int r = lane & 15, hi = lane >> 4;
  const __bf16* p = A + (size_t)r * lda + hi * 8;
  union { U4 q[2]; v16bf v; } u;
  u.q[0] = *reinterpret_cast<const U4*>(p);
  u.q[1] = *reinterpret_cast<const U4*>(p + 16);
  return u.v;
}
// B fragment: 32(K) x 16(N), B[k][n] = W[n][k] (weight row-major N x K).
// Lane l: column n = l&15; lanes 0-15 hold K {0..7,16..23}, lanes 16-31 K {8..15,24..31}.
DEV v16bf load_b(const __bf16* W, int ldb, int lane) {
  const int n = lane & 15, ko = (lane >> 4) * 8;
  const __bf16* p = W + (size_t)n * ldb + ko;
  union { U4 q[2]; v16bf v; } u;
  u.q[0] = *reinterpret_cast<const U4*>(p);
  u.q[1] = *reinterpret_cast<const U4*>(p + 16);
  return u.v;
}

enum { EPI_F32 = 0, EPI_F32_BF16 = 1, EPI_SOFTPLUS = 2, EPI_ACC = 3 };

// C[M,N] = A[MxK] * W[NxK]^T.
// Block = 128 thr = 4 waves arranged 2x2; wave tile = (16*MT) x (16*NT).
// <4,2>: 8 accumulators (64 VGPR) + 6 fragments/k-step (48 VGPR) -> no spill,
// 21 FLOP/byte from L2 (vs 16 at <2,2>).
template <int EPI, int MT, int NT>
__global__ __launch_bounds__(128, 1) void
gemm_nt(const __bf16* __restrict__ A, const __bf16* __restrict__ W,
        float* __restrict__ C, __bf16* __restrict__ Cb,
        const float* __restrict__ bias,
        int K, int lda, int ldb, int ldc) {
  const int lane = threadIdx.x & 31;
  const int wave = threadIdx.x >> 5;  // 0..3
  const int m0 = blockIdx.y * (32 * MT) + (wave >> 1) * (16 * MT);
  const int n0 = blockIdx.x * (32 * NT) + (wave & 1) * (16 * NT);

  v8f acc[MT][NT];
  #pragma unroll
  for (int mi = 0; mi < MT; ++mi)
    #pragma unroll
    for (int ni = 0; ni < NT; ++ni) acc[mi][ni] = {};

  for (int k = 0; k < K; k += 32) {
    v16bf a[MT];
    #pragma unroll
    for (int mi = 0; mi < MT; ++mi)
      a[mi] = load_a(A + (size_t)(m0 + 16 * mi) * lda + k, lda, lane);
    #pragma unroll
    for (int ni = 0; ni < NT; ++ni) {
      v16bf b = load_b(W + (size_t)(n0 + 16 * ni) * ldb + k, ldb, lane);
      #pragma unroll
      for (int mi = 0; mi < MT; ++mi)
        acc[mi][ni] = __builtin_amdgcn_wmma_f32_16x16x32_bf16(
            false, a[mi], false, b, (short)0, acc[mi][ni], false, false);
    }
  }

  // C/D layout: lane n = lane&15; VGPR j -> row j (lanes 0-15) / 8+j (lanes 16-31)
  const int colL = lane & 15;
  const int rowL = (lane >> 4) * 8;

  #pragma unroll
  for (int mi = 0; mi < MT; ++mi) {
    #pragma unroll
    for (int ni = 0; ni < NT; ++ni) {
      const int rbase = m0 + 16 * mi + rowL;
      const int col   = n0 + 16 * ni + colL;
      #pragma unroll
      for (int j = 0; j < 8; ++j) {
        float v = acc[mi][ni][j];
        size_t idx = (size_t)(rbase + j) * ldc + col;
        if constexpr (EPI == EPI_F32) {
          C[idx] = v;
        } else if constexpr (EPI == EPI_F32_BF16) {
          C[idx] = v; Cb[idx] = to_bf16(v);
        } else if constexpr (EPI == EPI_SOFTPLUS) {
          float t = v + bias[col];
          C[idx] = (t > 20.f) ? t : log1pf(__expf(t));
        } else {  // EPI_ACC
          C[idx] += v;
        }
      }
    }
  }
}

// ---------------- LayerNorm (row = 1024 elems, 256 threads x 4) --------------
__global__ void ln1_kernel(const float* __restrict__ x, const float* __restrict__ w,
                           const float* __restrict__ b, float* __restrict__ xn,
                           __bf16* __restrict__ xnb) {
  const int row = blockIdx.x, tid = threadIdx.x;
  const float* xr = x + (size_t)row * cfg::D;
  __shared__ float red[256];
  float v[4]; float s = 0.f;
  #pragma unroll
  for (int i = 0; i < 4; ++i) { v[i] = xr[tid + i * 256]; s += v[i]; }
  red[tid] = s; __syncthreads();
  for (int off = 128; off > 0; off >>= 1) {
    if (tid < off) red[tid] += red[tid + off];
    __syncthreads();
  }
  const float mu = red[0] * (1.f / cfg::D);
  __syncthreads();
  float sq = 0.f;
  #pragma unroll
  for (int i = 0; i < 4; ++i) { float d = v[i] - mu; sq += d * d; }
  red[tid] = sq; __syncthreads();
  for (int off = 128; off > 0; off >>= 1) {
    if (tid < off) red[tid] += red[tid + off];
    __syncthreads();
  }
  const float rs = rsqrtf(red[0] * (1.f / cfg::D) + cfg::EPS);
  #pragma unroll
  for (int i = 0; i < 4; ++i) {
    int c = tid + i * 256;
    float y = (v[i] - mu) * rs * w[c] + b[c];
    xn[(size_t)row * cfg::D + c] = y;
    xnb[(size_t)row * cfg::D + c] = to_bf16(y);
  }
}

__global__ void ln2_res_kernel(const float* __restrict__ xs, const float* __restrict__ w,
                               const float* __restrict__ b, const float* __restrict__ xn,
                               float* __restrict__ out) {
  const int row = blockIdx.x, tid = threadIdx.x;
  const float* xr = xs + (size_t)row * cfg::D;
  __shared__ float red[256];
  float v[4]; float s = 0.f;
  #pragma unroll
  for (int i = 0; i < 4; ++i) { v[i] = xr[tid + i * 256]; s += v[i]; }
  red[tid] = s; __syncthreads();
  for (int off = 128; off > 0; off >>= 1) {
    if (tid < off) red[tid] += red[tid + off];
    __syncthreads();
  }
  const float mu = red[0] * (1.f / cfg::D);
  __syncthreads();
  float sq = 0.f;
  #pragma unroll
  for (int i = 0; i < 4; ++i) { float d = v[i] - mu; sq += d * d; }
  red[tid] = sq; __syncthreads();
  for (int off = 128; off > 0; off >>= 1) {
    if (tid < off) red[tid] += red[tid + off];
    __syncthreads();
  }
  const float rs = rsqrtf(red[0] * (1.f / cfg::D) + cfg::EPS);
  #pragma unroll
  for (int i = 0; i < 4; ++i) {
    int c = tid + i * 256;
    float y = (v[i] - mu) * rs * w[c] + b[c];
    out[(size_t)row * cfg::D + c] = y + xn[(size_t)row * cfg::D + c];
  }
}

// --------- depthwise causal conv (K=4) + SiLU; reverse=1 flips time --------
__global__ void conv_silu_kernel(const float* __restrict__ xz, const float* __restrict__ cw,
                                 const float* __restrict__ cb, float* __restrict__ xc,
                                 __bf16* __restrict__ xcb, int reverse) {
  const int idx = blockIdx.x * blockDim.x + threadIdx.x;
  if (idx >= cfg::M * cfg::E) return;
  const int e  = idx % cfg::E;
  const int t  = (idx / cfg::E) % cfg::L;
  const int bz = idx / (cfg::E * cfg::L);
  float acc = cb[e];
  #pragma unroll
  for (int j = 0; j < cfg::KT; ++j) {
    const int tt = reverse ? (t + (cfg::KT - 1) - j) : (t - (cfg::KT - 1) + j);
    if (tt >= 0 && tt < cfg::L)
      acc += cw[e * cfg::KT + j] * xz[(size_t)(bz * cfg::L + tt) * cfg::E2 + e];
  }
  const float s = acc * sigmoidf_(acc);
  xc[idx]  = s;
  xcb[idx] = to_bf16(s);
}

// --------- selective scan: h[16] per lane, 4 lanes per channel e -----------
__global__ void scan_kernel(const float* __restrict__ xc, const float* __restrict__ dt,
                            const float* __restrict__ xdbl, const float* __restrict__ A_log,
                            float* __restrict__ ys, int reverse) {
  const int tid = threadIdx.x;
  const int g   = blockIdx.x * 256 + tid;
  const int e   = g >> 2;          // channel
  const int q   = g & 3;           // quarter of the N=64 state
  const int bz  = blockIdx.y;
  __shared__ float sBC[128];       // B (64) then C (64) for current timestep
  float h[16], Ar[16];
  #pragma unroll
  for (int i = 0; i < 16; ++i) {
    h[i]  = 0.f;
    Ar[i] = -__expf(A_log[(size_t)e * cfg::NS + q * 16 + i]);
  }
  for (int s = 0; s < cfg::L; ++s) {
    const int t = reverse ? (cfg::L - 1 - s) : s;
    const size_t row = (size_t)(bz * cfg::L + t);
    __syncthreads();
    if (tid < 128) sBC[tid] = xdbl[row * cfg::XW + cfg::R + tid];
    __syncthreads();
    const float u  = dt[row * cfg::E + e];
    const float dx = u * xc[row * cfg::E + e];
    float y = 0.f;
    #pragma unroll
    for (int i = 0; i < 16; ++i) {
      const int n = q * 16 + i;
      const float dA = __expf(u * Ar[i]);
      h[i] = dA * h[i] + dx * sBC[n];
      y += h[i] * sBC[64 + n];
    }
    y += __shfl_xor(y, 1);
    y += __shfl_xor(y, 2);
    if (q == 0) ys[row * cfg::E + e] = y;
  }
}

// --------- y = (ys + xc*Dp) * silu(z)  -> bf16 for final GEMM --------------
__global__ void yfuse_kernel(const float* __restrict__ ys, const float* __restrict__ xc,
                             const float* __restrict__ xz, const float* __restrict__ Dp,
                             __bf16* __restrict__ yb) {
  const int idx = blockIdx.x * blockDim.x + threadIdx.x;
  if (idx >= cfg::M * cfg::E) return;
  const int e = idx % cfg::E;
  const size_t row = (size_t)(idx / cfg::E);
  const float z = xz[row * cfg::E2 + cfg::E + e];
  const float y = (ys[idx] + xc[idx] * Dp[e]) * (z * sigmoidf_(z));
  yb[idx] = to_bf16(y);
}

__global__ void f32_to_bf16_kernel(const float* __restrict__ src, __bf16* __restrict__ dst, int n) {
  const int i = blockIdx.x * blockDim.x + threadIdx.x;
  if (i < n) dst[i] = to_bf16(src[i]);
}

extern "C" void kernel_launch(void* const* d_in, const int* in_sizes, int n_in,
                              void* d_out, int out_size, void* d_ws, size_t ws_size,
                              hipStream_t stream) {
  using namespace cfg;
  (void)in_sizes; (void)n_in; (void)out_size; (void)ws_size;
  const float* x     = (const float*)d_in[0];
  const float* ln1_w = (const float*)d_in[1];
  const float* ln1_b = (const float*)d_in[2];
  const float* ln2_w = (const float*)d_in[3];
  const float* ln2_b = (const float*)d_in[4];
  const float* p[2][9];
  for (int dd = 0; dd < 2; ++dd)
    for (int k = 0; k < 9; ++k) p[dd][k] = (const float*)d_in[5 + dd * 9 + k];
  // per-direction order: W_in, conv_w, conv_b, W_xp, W_dt, dt_b, A_log, D, W_out

  char* ws = (char*)d_ws;
  auto alloc = [&](size_t bytes) -> void* {
    void* r = (void*)ws;
    ws += (bytes + 255) & ~(size_t)255;
    return r;
  };
  float*  xn    = (float*) alloc((size_t)M * D  * 4);
  __bf16* xnb   = (__bf16*)alloc((size_t)M * D  * 2);
  float*  xz    = (float*) alloc((size_t)M * E2 * 4);
  float*  xc    = (float*) alloc((size_t)M * E  * 4);
  __bf16* xcb   = (__bf16*)alloc((size_t)M * E  * 2);
  float*  xdbl  = (float*) alloc((size_t)M * XW * 4);
  __bf16* xdblb = (__bf16*)alloc((size_t)M * XW * 2);
  float*  dtb_  = (float*) alloc((size_t)M * E  * 4);
  float*  ysb   = (float*) alloc((size_t)M * E  * 4);
  __bf16* yb    = (__bf16*)alloc((size_t)M * E  * 2);
  float*  xsum  = (float*) alloc((size_t)M * D  * 4);
  __bf16* wInB  = (__bf16*)alloc((size_t)E2 * D * 2);
  __bf16* wXpB  = (__bf16*)alloc((size_t)XW * E * 2);
  __bf16* wDtB  = (__bf16*)alloc((size_t)E  * R * 2);
  __bf16* wOutB = (__bf16*)alloc((size_t)D  * E * 2);

  ln1_kernel<<<M, 256, 0, stream>>>(x, ln1_w, ln1_b, xn, xnb);

  const int cvt = 256;
  for (int dd = 0; dd < 2; ++dd) {
    const float* W_in  = p[dd][0];
    const float* cw    = p[dd][1];
    const float* cb    = p[dd][2];
    const float* W_xp  = p[dd][3];
    const float* W_dt  = p[dd][4];
    const float* dt_b  = p[dd][5];
    const float* A_log = p[dd][6];
    const float* Dp    = p[dd][7];
    const float* W_out = p[dd][8];

    f32_to_bf16_kernel<<<(E2 * D + cvt - 1) / cvt, cvt, 0, stream>>>(W_in,  wInB,  E2 * D);
    f32_to_bf16_kernel<<<(XW * E + cvt - 1) / cvt, cvt, 0, stream>>>(W_xp,  wXpB,  XW * E);
    f32_to_bf16_kernel<<<(E  * R + cvt - 1) / cvt, cvt, 0, stream>>>(W_dt,  wDtB,  E  * R);
    f32_to_bf16_kernel<<<(D  * E + cvt - 1) / cvt, cvt, 0, stream>>>(W_out, wOutB, D  * E);

    // xz = xn @ W_in^T  (M x E2, K=D), 128x64 block tile
    gemm_nt<EPI_F32, 4, 2><<<dim3(E2 / 64, M / 128), 128, 0, stream>>>(
        xnb, wInB, xz, nullptr, nullptr, D, D, D, E2);
    // xc = silu(causal_conv(xp) + cb)   (direction-aware)
    conv_silu_kernel<<<(M * E) / 256, 256, 0, stream>>>(xz, cw, cb, xc, xcb, dd);
    // x_dbl = xc @ W_xp^T  (M x 192, K=E): N=192 -> 64x64 block tile
    gemm_nt<EPI_F32_BF16, 2, 2><<<dim3(XW / 64, M / 64), 128, 0, stream>>>(
        xcb, wXpB, xdbl, xdblb, nullptr, E, E, E, XW);
    // dt = softplus(dt_lo @ W_dt^T + dt_b)  (M x E, K=R)
    gemm_nt<EPI_SOFTPLUS, 4, 2><<<dim3(E / 64, M / 128), 128, 0, stream>>>(
        xdblb, wDtB, dtb_, nullptr, dt_b, R, XW, R, E);
    // selective scan (direction-aware)
    scan_kernel<<<dim3(E * 4 / 256, B), 256, 0, stream>>>(xc, dtb_, xdbl, A_log, ysb, dd);
    // gating
    yfuse_kernel<<<(M * E) / 256, 256, 0, stream>>>(ysb, xc, xz, Dp, yb);
    // branch output: y @ W_out^T (M x D, K=E); dir1 stores, dir2 accumulates
    if (dd == 0)
      gemm_nt<EPI_F32, 4, 2><<<dim3(D / 64, M / 128), 128, 0, stream>>>(
          yb, wOutB, xsum, nullptr, nullptr, E, E, E, D);
    else
      gemm_nt<EPI_ACC, 4, 2><<<dim3(D / 64, M / 128), 128, 0, stream>>>(
          yb, wOutB, xsum, nullptr, nullptr, E, E, E, D);
  }

  ln2_res_kernel<<<M, 256, 0, stream>>>(xsum, ln2_w, ln2_b, xn, (float*)d_out);
}
